// GeoER_1709396984342
// MI455X (gfx1250) — matrix operand order repaired and checked by reading, not compiled, not run on hardware
//
#include <hip/hip_runtime.h>
#include <hip/hip_bf16.h>

typedef __attribute__((ext_vector_type(16))) _Float16 v16h;
typedef __attribute__((ext_vector_type(8)))  float    v8f;

#define NEGV (-1e30f)

// Pack two f32 into one dword of two f16 with a single CDNA5 VALU op.
// Non-volatile asm: scheduler may reorder/CSE freely; RNE rounding (default mode).
__device__ __forceinline__ unsigned pk2(float a, float b) {
    unsigned r;
    asm("v_cvt_pk_f16_f32 %0, %1, %2" : "=v"(r) : "v"(a), "v"(b));
    return r;
}

// Build one A-fragment lane slice (8 dwords = 16 f16) from an f32 row.
// 16-bit A layout: lane hi-half offset +8 in K, dwords 0-3 -> K 0..7, dwords 4-7 -> K 16..23.
__device__ __forceinline__ void load_a_lane(const float* __restrict__ arow,
                                            int kbase, int hi, unsigned* u) {
    const float4* p0 = (const float4*)(arow + kbase + 8 * hi);
    const float4* p1 = (const float4*)(arow + kbase + 16 + 8 * hi);
    float4 x0 = p0[0], x1 = p0[1];
    float4 y0 = p1[0], y1 = p1[1];
    u[0] = pk2(x0.x, x0.y); u[1] = pk2(x0.z, x0.w);
    u[2] = pk2(x1.x, x1.y); u[3] = pk2(x1.z, x1.w);
    u[4] = pk2(y0.x, y0.y); u[5] = pk2(y0.z, y0.w);
    u[6] = pk2(y1.x, y1.y); u[7] = pk2(y1.z, y1.w);
}

// ---------------------------------------------------------------------------
// Pack a row-major [K,N] f32 matrix into f16 WMMA B-fragments (32x16 tiles).
//   lane l: n = nt*16 + (l&15), hi = l>>4; dword d holds K = kt*32 + 8*hi + 2*(d&3) + 16*(d>>2), +1
// ---------------------------------------------------------------------------
__global__ void pack_b_frags(const float* __restrict__ W, unsigned* __restrict__ out,
                             int K, int N) {
    int NT = N >> 4;
    int total = (K >> 5) * NT * 32 * 8;
    for (int i = blockIdx.x * blockDim.x + threadIdx.x; i < total;
         i += gridDim.x * blockDim.x) {
        int d    = i & 7;
        int lane = (i >> 3) & 31;
        int f    = i >> 8;
        int nt   = f % NT;
        int kt   = f / NT;
        int n    = nt * 16 + (lane & 15);
        int hi   = lane >> 4;
        int kk   = kt * 32 + 8 * hi + 2 * (d & 3) + 16 * (d >> 2);
        out[i] = pk2(W[(size_t)kk * N + n], W[(size_t)(kk + 1) * N + n]);
    }
}

// ---------------------------------------------------------------------------
// Attention side kernel: one block per (batch b, side s). 256 threads = 8 waves.
// A is a logical 128x768 matrix: rows 0..99 = neighbors, 100..111 padding
// (clamped to row 0; killed by att==0), row 112 = node (gives wn), 113..127
// padding (discarded).  Each wave owns one 16-wide N slice and runs 4 M-tile
// pairs with dual accumulators sharing each B fragment.
// ---------------------------------------------------------------------------
#define WSTR 129

__device__ __forceinline__ const float* attn_rowptr(const float* __restrict__ nbase,
                                                    const float* __restrict__ node,
                                                    int rl) {
    if (rl == 112) return node;
    return nbase + (size_t)(rl < 100 ? rl : 0) * 768;
}

__global__ __launch_bounds__(256) void attn_kernel(
    const float* __restrict__ node1, const float* __restrict__ node2,
    const float* __restrict__ neigh1, const float* __restrict__ neigh2,
    const float* __restrict__ dist1, const float* __restrict__ dist2,
    const int* __restrict__ len1, const int* __restrict__ len2,
    const float* __restrict__ bw, const float* __restrict__ Wa,
    const float* __restrict__ ba, const float* __restrict__ Wdb,
    const float* __restrict__ bdb, const unsigned* __restrict__ WwF16,
    float* __restrict__ simbuf) {
    __shared__ float wng[112 * WSTR];
    __shared__ float wn[128];
    __shared__ float att[128];
    __shared__ float red[256];

    const int b = blockIdx.x;
    const int s = blockIdx.y;
    const float* node  = (s ? node2  : node1) + (size_t)b * 768;
    const float* nbase = (s ? neigh2 : neigh1) + (size_t)b * 100 * 768;
    const float* dist  = s ? dist2  : dist1;
    const int    len   = (s ? len2 : len1)[b];
    const int t = threadIdx.x;

    const int wave = t >> 5, lane = t & 31;
    const int m = lane & 15, hi = lane >> 4;
    const int col = wave * 16 + m;
    const float bias = bw[col];

    // ---- wng (and wn as row 112) via WMMA, 4 M-tile pairs per wave ----
    for (int mtp = 0; mtp < 4; ++mtp) {
        v8f c0 = {}, c1 = {};
        const float* arow0 = attn_rowptr(nbase, node, mtp * 32 + m);
        const float* arow1 = attn_rowptr(nbase, node, mtp * 32 + 16 + m);
        for (int kt = 0; kt < 24; ++kt) {
            union { v16h v; unsigned u[8]; } a0, a1, bm;
            load_a_lane(arow0, kt * 32, hi, a0.u);
            load_a_lane(arow1, kt * 32, hi, a1.u);
            const unsigned* bf = WwF16 + ((size_t)(kt * 8 + wave) * 32 + lane) * 8;
            if (kt + 1 < 24) __builtin_prefetch((const void*)(bf + 2048), 0, 1);
            uint4 q0 = ((const uint4*)bf)[0];
            uint4 q1 = ((const uint4*)bf)[1];
            bm.u[0] = q0.x; bm.u[1] = q0.y; bm.u[2] = q0.z; bm.u[3] = q0.w;
            bm.u[4] = q1.x; bm.u[5] = q1.y; bm.u[6] = q1.z; bm.u[7] = q1.w;
            c0 = __builtin_amdgcn_wmma_f32_16x16x32_f16(false, a0.v, false, bm.v,
                                                        (short)0, c0, false, false);
            c1 = __builtin_amdgcn_wmma_f32_16x16x32_f16(false, a1.v, false, bm.v,
                                                        (short)0, c1, false, false);
        }
#pragma unroll
        for (int r = 0; r < 8; ++r) {
            const int row0 = mtp * 32 + r + 8 * hi;
            wng[(size_t)row0 * WSTR + col] = c0[r] + bias;
            const int row1 = row0 + 16;
            if (row1 < 112)       wng[(size_t)row1 * WSTR + col] = c1[r] + bias;
            else if (row1 == 112) wn[col] = c1[r] + bias;
        }
    }
    __syncthreads();

    // ---- waWn = wn . Wa[0:128]  (deterministic tree reduce) ----
    red[t] = (t < 128) ? wn[t] * Wa[t] : 0.f;
    __syncthreads();
    for (int off = 128; off; off >>= 1) {
        if (t < off) red[t] += red[t + off];
        __syncthreads();
    }
    const float waWn = red[0] + ba[0];
    const float wdb  = Wdb[0];
    const float bd   = bdb[0];
    __syncthreads();

    // ---- scores ----
    float sc = NEGV;
    if (t < 112) {
        float s2 = 0.f;
        for (int j = 0; j < 128; ++j)
            s2 = fmaf(wng[(size_t)t * WSTR + j], Wa[128 + j], s2);
        if (t < len) {
            float v = waWn + s2;
            v  = v > 0.f ? v : 0.01f * v;            // leaky_relu(..., 0.01)
            sc = v + dist[(size_t)b * 100 + t] * wdb + bd;
        }
    }

    // ---- softmax over 112 (padded) neighbors ----
    red[t] = sc;
    __syncthreads();
    for (int off = 128; off; off >>= 1) {
        if (t < off) red[t] = fmaxf(red[t], red[t + off]);
        __syncthreads();
    }
    const float mx = red[0];
    __syncthreads();
    const float e = (t < 112) ? expf(sc - mx) : 0.f;
    red[t] = e;
    __syncthreads();
    for (int off = 128; off; off >>= 1) {
        if (t < off) red[t] += red[t + off];
        __syncthreads();
    }
    const float denom = red[0];
    __syncthreads();
    if (t < 112) att[t] = e / denom;  // att[len..111] == 0 exactly
    __syncthreads();

    // ---- ctx & sim ----
    if (t < 128) {
        float cx = 0.f;
        for (int n = 0; n < 112; ++n)
            cx = fmaf(att[n], wng[(size_t)n * WSTR + t], cx);
        simbuf[(size_t)b * 256 + s * 128 + t] = cx * wn[t];
    }
}

// ---------------------------------------------------------------------------
// h = [pooled | coord | relu(sim) @ Wn + bn]   ->   hbuf [B,1152]
// ---------------------------------------------------------------------------
__global__ __launch_bounds__(256) void assemble_kernel(
    const float* __restrict__ pooled, const float* __restrict__ x_coord,
    const float* __restrict__ simbuf, const float* __restrict__ Wn,
    const float* __restrict__ bn, const float* __restrict__ Wc,
    const float* __restrict__ bc, float* __restrict__ hbuf) {
    const int b = blockIdx.x, t = threadIdx.x;
    float* h = hbuf + (size_t)b * 1152;

    for (int i = t; i < 768; i += 256) h[i] = pooled[(size_t)b * 768 + i];

    const float xc = x_coord[b];
    if (t < 128) h[768 + t] = xc * Wc[t] + bc[t];

    float acc = bn[t];
    const float* sb = simbuf + (size_t)b * 256;
    for (int j = 0; j < 256; ++j) {
        float rv = sb[j];
        rv = rv > 0.f ? rv : 0.f;
        acc = fmaf(rv, Wn[(size_t)j * 256 + t], acc);
    }
    h[896 + t] = acc;
}

// ---------------------------------------------------------------------------
// h2 = gelu(h @ W1 + b1), exact erf GELU. 16 batch rows per block, WMMA.
// A staged once in LDS; each wave runs dual N-tiles sharing every A fragment.
// ---------------------------------------------------------------------------
__global__ __launch_bounds__(256) void mlp_kernel(
    const float* __restrict__ hbuf, const unsigned* __restrict__ W1F16,
    const float* __restrict__ b1, float* __restrict__ h2buf) {
    __shared__ unsigned aF[36 * 32 * 8];  // 36 K-tiles of A (16x32 f16 each)

    const int t  = threadIdx.x;
    const int b0 = blockIdx.x * 16;

    for (int slot = t; slot < 36 * 32; slot += 256) {
        const int kt = slot >> 5, lane = slot & 31;
        const float* row = hbuf + (size_t)(b0 + (lane & 15)) * 1152;
        load_a_lane(row, kt * 32, lane >> 4, aF + (size_t)slot * 8);
    }
    __syncthreads();

    const int wave = t >> 5, lane = t & 31;
    const int rowbase = b0 + 8 * (lane >> 4);

    // dual N-tile pairs: (base+wave, base+wave+8) for base in {0,16}
    for (int base = 0; base < 32; base += 16) {
        const int nt0 = base + wave, nt1 = base + wave + 8;
        v8f c0 = {}, c1 = {};
        for (int kt = 0; kt < 36; ++kt) {
            union { v16h v; unsigned u[8]; } a, bm0, bm1;
            const unsigned* ap = aF + (size_t)(kt * 32 + lane) * 8;
            uint4 d0 = ((const uint4*)ap)[0];
            uint4 d1 = ((const uint4*)ap)[1];
            a.u[0] = d0.x; a.u[1] = d0.y; a.u[2] = d0.z; a.u[3] = d0.w;
            a.u[4] = d1.x; a.u[5] = d1.y; a.u[6] = d1.z; a.u[7] = d1.w;
            const unsigned* bf0 = W1F16 + ((size_t)(kt * 36 + nt0) * 32 + lane) * 8;
            const unsigned* bf1 = W1F16 + ((size_t)(kt * 36 + nt1) * 32 + lane) * 8;
            uint4 q0 = ((const uint4*)bf0)[0];
            uint4 q1 = ((const uint4*)bf0)[1];
            bm0.u[0] = q0.x; bm0.u[1] = q0.y; bm0.u[2] = q0.z; bm0.u[3] = q0.w;
            bm0.u[4] = q1.x; bm0.u[5] = q1.y; bm0.u[6] = q1.z; bm0.u[7] = q1.w;
            uint4 r0 = ((const uint4*)bf1)[0];
            uint4 r1 = ((const uint4*)bf1)[1];
            bm1.u[0] = r0.x; bm1.u[1] = r0.y; bm1.u[2] = r0.z; bm1.u[3] = r0.w;
            bm1.u[4] = r1.x; bm1.u[5] = r1.y; bm1.u[6] = r1.z; bm1.u[7] = r1.w;
            c0 = __builtin_amdgcn_wmma_f32_16x16x32_f16(false, a.v, false, bm0.v,
                                                        (short)0, c0, false, false);
            c1 = __builtin_amdgcn_wmma_f32_16x16x32_f16(false, a.v, false, bm1.v,
                                                        (short)0, c1, false, false);
        }
#pragma unroll
        for (int r = 0; r < 8; ++r) {
            const size_t orow = (size_t)(rowbase + r) * 576;
            {
                const int col = nt0 * 16 + (lane & 15);
                float v = c0[r] + b1[col];
                v = 0.5f * v * (1.f + erff(v * 0.70710678118654752f));
                h2buf[orow + col] = v;
            }
            {
                const int col = nt1 * 16 + (lane & 15);
                float v = c1[r] + b1[col];
                v = 0.5f * v * (1.f + erff(v * 0.70710678118654752f));
                h2buf[orow + col] = v;
            }
        }
    }
    // remaining singleton N-tiles 32..35 for waves 0..3
    if (wave < 4) {
        const int nt = 32 + wave;
        v8f c = {};
        for (int kt = 0; kt < 36; ++kt) {
            union { v16h v; unsigned u[8]; } a, bm;
            const unsigned* ap = aF + (size_t)(kt * 32 + lane) * 8;
            uint4 d0 = ((const uint4*)ap)[0];
            uint4 d1 = ((const uint4*)ap)[1];
            a.u[0] = d0.x; a.u[1] = d0.y; a.u[2] = d0.z; a.u[3] = d0.w;
            a.u[4] = d1.x; a.u[5] = d1.y; a.u[6] = d1.z; a.u[7] = d1.w;
            const unsigned* bf = W1F16 + ((size_t)(kt * 36 + nt) * 32 + lane) * 8;
            uint4 q0 = ((const uint4*)bf)[0];
            uint4 q1 = ((const uint4*)bf)[1];
            bm.u[0] = q0.x; bm.u[1] = q0.y; bm.u[2] = q0.z; bm.u[3] = q0.w;
            bm.u[4] = q1.x; bm.u[5] = q1.y; bm.u[6] = q1.z; bm.u[7] = q1.w;
            c = __builtin_amdgcn_wmma_f32_16x16x32_f16(false, a.v, false, bm.v,
                                                       (short)0, c, false, false);
        }
        const int col = nt * 16 + (lane & 15);
        const float bias = b1[col];
#pragma unroll
        for (int r = 0; r < 8; ++r) {
            float v = c[r] + bias;
            v = 0.5f * v * (1.f + erff(v * 0.70710678118654752f));
            h2buf[(size_t)(rowbase + r) * 576 + col] = v;
        }
    }
}

// ---------------------------------------------------------------------------
// logits = h2 @ W2 + b2 ; out = log_softmax(logits)
// ---------------------------------------------------------------------------
__global__ void head_kernel(const float* __restrict__ h2buf,
                            const float* __restrict__ W2,
                            const float* __restrict__ b2,
                            float* __restrict__ out, int B) {
    const int b = blockIdx.x * blockDim.x + threadIdx.x;
    if (b >= B) return;
    const float* h = h2buf + (size_t)b * 576;
    float a0 = b2[0], a1 = b2[1];
    for (int k = 0; k < 576; ++k) {
        const float v = h[k];
        a0 = fmaf(v, W2[2 * k], a0);
        a1 = fmaf(v, W2[2 * k + 1], a1);
    }
    const float mx  = fmaxf(a0, a1);
    const float lse = mx + logf(expf(a0 - mx) + expf(a1 - mx));
    out[2 * b]     = a0 - lse;
    out[2 * b + 1] = a1 - lse;
}

// ---------------------------------------------------------------------------
extern "C" void kernel_launch(void* const* d_in, const int* in_sizes, int n_in,
                              void* d_out, int out_size, void* d_ws, size_t ws_size,
                              hipStream_t stream) {
    const int B = 1024;

    const float* pooled  = (const float*)d_in[0];
    const float* x_coord = (const float*)d_in[1];
    const float* node1   = (const float*)d_in[2];
    const float* node2   = (const float*)d_in[3];
    const float* neigh1  = (const float*)d_in[4];
    const float* neigh2  = (const float*)d_in[5];
    const float* dist1   = (const float*)d_in[6];
    const float* dist2   = (const float*)d_in[7];
    const int*   len1    = (const int*)d_in[8];
    const int*   len2    = (const int*)d_in[9];
    const float* Ww  = (const float*)d_in[10];
    const float* bw  = (const float*)d_in[11];
    const float* Wa  = (const float*)d_in[12];
    const float* ba  = (const float*)d_in[13];
    const float* Wdb = (const float*)d_in[14];
    const float* bdb = (const float*)d_in[15];
    const float* Wn  = (const float*)d_in[16];
    const float* bn  = (const float*)d_in[17];
    const float* Wc  = (const float*)d_in[18];
    const float* bc  = (const float*)d_in[19];
    const float* W1  = (const float*)d_in[20];
    const float* b1  = (const float*)d_in[21];
    const float* W2  = (const float*)d_in[22];
    const float* b2  = (const float*)d_in[23];
    float* out = (float*)d_out;

    // workspace carve-up
    char* ws = (char*)d_ws;
    unsigned* WwF16 = (unsigned*)ws;                ws += (size_t)24 * 8  * 256 * 4;  // 192 KB
    unsigned* W1F16 = (unsigned*)ws;                ws += (size_t)36 * 36 * 256 * 4;  // 1.27 MB
    float* simbuf = (float*)ws;                     ws += (size_t)B * 256 * 4;
    float* hbuf   = (float*)ws;                     ws += (size_t)B * 1152 * 4;
    float* h2buf  = (float*)ws;                     ws += (size_t)B * 576 * 4;

    // 1) pack weights to f16 WMMA fragments
    pack_b_frags<<<192, 256, 0, stream>>>(Ww, WwF16, 768, 128);
    pack_b_frags<<<1296, 256, 0, stream>>>(W1, W1F16, 1152, 576);

    // 2) attention sides (grid.y = side); wn folded in as A-row 112
    attn_kernel<<<dim3(B, 2), 256, 0, stream>>>(node1, node2, neigh1, neigh2,
                                                dist1, dist2, len1, len2,
                                                bw, Wa, ba, Wdb, bdb,
                                                WwF16, simbuf);

    // 3) assemble h = [pooled | coord | xneigh]
    assemble_kernel<<<B, 256, 0, stream>>>(pooled, x_coord, simbuf, Wn, bn, Wc, bc, hbuf);

    // 4) h2 = gelu(h @ W1 + b1)   (WMMA)
    mlp_kernel<<<B / 16, 256, 0, stream>>>(hbuf, W1F16, b1, h2buf);

    // 5) logits + log_softmax
    head_kernel<<<(B + 255) / 256, 256, 0, stream>>>(h2buf, W2, b2, out, B);
}
